// PatchedAttentionTransformerDecoderLayer_20023137534418
// MI455X (gfx1250) — matrix operand order repaired and checked by reading, not compiled, not run on hardware
//
#include <hip/hip_runtime.h>
#include <hip/hip_bf16.h>
#include <stdint.h>

typedef __attribute__((ext_vector_type(16))) __bf16 v16bf;
typedef __attribute__((ext_vector_type(8)))  float  v8f;
typedef __attribute__((ext_vector_type(2)))  float  v2f;

#define DM_C   1024
#define P_C    256
#define HD_C   64
#define T_C    64
#define S_C    2048
#define H_C    16
#define B_C    16

// ---------------------------------------------------------------------------
// Async copy helpers (gfx1250 async-to-LDS path, tracked by ASYNCcnt)
// ---------------------------------------------------------------------------
__device__ __forceinline__ void async_b128(unsigned lds_addr, const void* gaddr) {
  asm volatile("global_load_async_to_lds_b128 %0, %1, off"
               :: "v"(lds_addr), "v"((unsigned long long)(uintptr_t)gaddr)
               : "memory");
}
__device__ __forceinline__ void wait_async0() {
  asm volatile("s_wait_asynccnt 0" ::: "memory");
}
__device__ __forceinline__ unsigned lds_off(const void* p) {
  // generic->LDS: low 32 bits of the flat address are the LDS byte offset
  return (unsigned)(uintptr_t)p;
}

// ---------------------------------------------------------------------------
// Weight repack: W[K,N] f32  ->  WT_hi[N,K], WT_lo[N,K] bf16 (hi/lo split)
// ---------------------------------------------------------------------------
__global__ __launch_bounds__(256) void repack_weight(
    const float* __restrict__ W, __bf16* __restrict__ Th, __bf16* __restrict__ Tl,
    int K, int N)
{
  int i = blockIdx.x * 256 + threadIdx.x;
  if (i >= K * N) return;
  int k = i / N;
  int n = i - k * N;
  float x = W[i];
  __bf16 h = (__bf16)x;
  float r = x - (float)h;
  Th[(size_t)n * K + k] = h;
  Tl[(size_t)n * K + k] = (__bf16)r;
}

// ---------------------------------------------------------------------------
// Generic bf16x3 GEMM:  C[M,N] = act(A[M,K] @ W[K,N] + bias)
// Double-buffered LDS; weight panels staged with async-to-LDS b128.
// Block = 4 waves, 64x64 tile.  M%64==0, N%64==0, K%32==0.
// ---------------------------------------------------------------------------
template<bool RELU>
__global__ __launch_bounds__(128) void gemm_bf16x3(
    const float*  __restrict__ A,    // [M,K]
    const __bf16* __restrict__ WTh,  // [N,K]
    const __bf16* __restrict__ WTl,  // [N,K]
    const float*  __restrict__ bias, // [N]
    float* __restrict__ C,           // [M,N]
    int M, int N, int K)
{
  __shared__ __bf16 Ahs[2][64][32];
  __shared__ __bf16 Als[2][64][32];
  __shared__ __bf16 Bhs[2][64][32];
  __shared__ __bf16 Bls[2][64][32];

  const int tid  = threadIdx.x;
  const int lane = tid & 31;
  const int w    = tid >> 5;
  const int half = lane >> 4;
  const int l15  = lane & 15;
  const int m0   = blockIdx.x * 64;
  const int n0   = blockIdx.y * 64;

  v8f acc[4];
#pragma unroll
  for (int nt = 0; nt < 4; ++nt)
#pragma unroll
    for (int r = 0; r < 8; ++r) acc[nt][r] = 0.0f;

  // --- staging helpers (uniform trip counts; all lanes active) -------------
  auto issue_w_panel = [&](int buf, int k0) {
#pragma unroll
    for (int it = 0; it < 2; ++it) {          // 256 b128 segs / 128 threads
      int i = tid + it * 128;
      int r = i >> 2, c = (i & 3) * 8;        // 8 bf16 = 16 bytes
      const __bf16* gh = WTh + (size_t)(n0 + r) * K + k0 + c;
      const __bf16* gl = WTl + (size_t)(n0 + r) * K + k0 + c;
      async_b128(lds_off(&Bhs[buf][r][c]), gh);
      async_b128(lds_off(&Bls[buf][r][c]), gl);
    }
  };
  float4 areg[4];
  auto load_a_panel = [&](int k0) {
#pragma unroll
    for (int it = 0; it < 4; ++it) {          // 512 float4 segs / 128 threads
      int i = tid + it * 128;
      int r = i >> 3, c = (i & 7) * 4;
      areg[it] = *(const float4*)(A + (size_t)(m0 + r) * K + k0 + c);
    }
  };
  auto store_a_panel = [&](int buf) {
#pragma unroll
    for (int it = 0; it < 4; ++it) {
      int i = tid + it * 128;
      int r = i >> 3, c = (i & 7) * 4;
      float vv[4] = {areg[it].x, areg[it].y, areg[it].z, areg[it].w};
#pragma unroll
      for (int j = 0; j < 4; ++j) {
        __bf16 h = (__bf16)vv[j];
        Ahs[buf][r][c + j] = h;
        Als[buf][r][c + j] = (__bf16)(vv[j] - (float)h);
      }
    }
  };

  // prologue: stage panel 0
  issue_w_panel(0, 0);
  load_a_panel(0);
  store_a_panel(0);
  wait_async0();
  __syncthreads();

  int buf = 0;
  for (int k0 = 0; k0 < K; k0 += 32) {
    const int nxt = buf ^ 1;
    const bool more = (k0 + 32 < K);
    if (more) {
      issue_w_panel(nxt, k0 + 32);   // async, overlaps compute
      load_a_panel(k0 + 32);         // global loads overlap compute
    }

    // A fragment: row m = w*16 + l15 ; K(j) = j + 8*(j>=8) + 8*half
    const int am = w * 16 + l15;
    v16bf fAh, fAl;
#pragma unroll
    for (int j = 0; j < 8; ++j) {
      fAh[j]     = Ahs[buf][am][half * 8 + j];
      fAl[j]     = Als[buf][am][half * 8 + j];
      fAh[8 + j] = Ahs[buf][am][16 + half * 8 + j];
      fAl[8 + j] = Als[buf][am][16 + half * 8 + j];
    }
#pragma unroll
    for (int nt = 0; nt < 4; ++nt) {
      const int bn = nt * 16 + l15;
      v16bf fBh, fBl;
#pragma unroll
      for (int j = 0; j < 16; ++j) {
        int kk = (j < 8) ? half * 8 + j : 16 + half * 8 + (j - 8);     // B: K(j) = j + 8*(j>=8) + 8*half
        fBh[j] = Bhs[buf][bn][kk];
        fBl[j] = Bls[buf][bn][kk];
      }
      acc[nt] = __builtin_amdgcn_wmma_f32_16x16x32_bf16(false, fAh, false, fBh,
                                                        (short)0, acc[nt], false, false);
      acc[nt] = __builtin_amdgcn_wmma_f32_16x16x32_bf16(false, fAh, false, fBl,
                                                        (short)0, acc[nt], false, false);
      acc[nt] = __builtin_amdgcn_wmma_f32_16x16x32_bf16(false, fAl, false, fBh,
                                                        (short)0, acc[nt], false, false);
    }

    if (more) store_a_panel(nxt);    // nxt was last read before the previous barrier: safe
    wait_async0();
    __syncthreads();
    buf = nxt;
  }

  // epilogue: C layout -> row = r + 8*half, col = l15
#pragma unroll
  for (int nt = 0; nt < 4; ++nt) {
    const int col = n0 + nt * 16 + l15;
    const float b = bias[col];
#pragma unroll
    for (int r = 0; r < 8; ++r) {
      int row = m0 + w * 16 + half * 8 + r;
      float v = acc[nt][r] + b;
      if (RELU) v = fmaxf(v, 0.0f);
      C[(size_t)row * N + col] = v;
    }
  }
}

// ---------------------------------------------------------------------------
// Flash attention per (b,h) block, exact f32 WMMA (16x16x4_f32).
// K/V chunks staged via async-to-LDS b128.
// Block = 4 waves (128 threads); wave w owns query rows [16w,16w+16).
// ---------------------------------------------------------------------------
__global__ __launch_bounds__(128) void attn_kernel(
    const float* __restrict__ Q,   // [B*T, DM]
    const float* __restrict__ Kp,  // [B*S, DM]
    const float* __restrict__ Vp,  // [B*S, DM]
    float* __restrict__ O)         // [B*T, DM]
{
  __shared__ float Kc[64][64];
  __shared__ float Vc[64][64];
  __shared__ float Ps[64][64];

  const int b = blockIdx.x / H_C;
  const int h = blockIdx.x % H_C;
  const int tid  = threadIdx.x;
  const int lane = tid & 31;
  const int w    = tid >> 5;
  const int half = lane >> 4;
  const int l15  = lane & 15;
  const float scale = 0.125f; // 1/sqrt(64)

  // Q as A-fragments (16 k-steps of K=4), pre-scaled
  v2f qA[16];
  {
    const int tq = w * 16 + l15;
    const float* qrow = Q + (size_t)(b * T_C + tq) * DM_C + h * HD_C;
#pragma unroll
    for (int ks = 0; ks < 16; ++ks) {
      qA[ks][0] = qrow[ks * 4 + half * 2 + 0] * scale;
      qA[ks][1] = qrow[ks * 4 + half * 2 + 1] * scale;
    }
  }

  float mrow[8], lrow[8];
  v8f oacc[4];
#pragma unroll
  for (int r = 0; r < 8; ++r) { mrow[r] = -INFINITY; lrow[r] = 0.0f; }
#pragma unroll
  for (int nt = 0; nt < 4; ++nt)
#pragma unroll
    for (int r = 0; r < 8; ++r) oacc[nt][r] = 0.0f;

  for (int s0 = 0; s0 < S_C; s0 += 64) {
    __syncthreads();   // prior chunk's LDS reads complete
    // stage K,V chunks [64 s x 64 d] via async DMA (16 b128 per thread)
    {
      const float* kb = Kp + (size_t)(b * S_C + s0) * DM_C + h * HD_C;
      const float* vb = Vp + (size_t)(b * S_C + s0) * DM_C + h * HD_C;
#pragma unroll
      for (int it = 0; it < 8; ++it) {        // 1024 b128 segs / 128 threads
        int i = tid + it * 128;
        int r = i >> 4, c = (i & 15) * 4;     // 4 floats = 16 bytes
        async_b128(lds_off(&Kc[r][c]), kb + (size_t)r * DM_C + c);
        async_b128(lds_off(&Vc[r][c]), vb + (size_t)r * DM_C + c);
      }
    }
    wait_async0();
    __syncthreads();

    // scores S = Q K^T  (contraction over d)
    v8f sacc[4];
#pragma unroll
    for (int nt = 0; nt < 4; ++nt)
#pragma unroll
      for (int r = 0; r < 8; ++r) sacc[nt][r] = 0.0f;

#pragma unroll
    for (int ks = 0; ks < 16; ++ks) {
      v2f a = qA[ks];
#pragma unroll
      for (int nt = 0; nt < 4; ++nt) {
        v2f bf;
        int s = nt * 16 + l15;                       // B: N = s
        bf[0] = Kc[s][ks * 4 + half * 2 + 0];        // B: K = d
        bf[1] = Kc[s][ks * 4 + half * 2 + 1];
        sacc[nt] = __builtin_amdgcn_wmma_f32_16x16x4_f32(false, a, false, bf,
                                                         (short)0, sacc[nt], false, false);
      }
    }

    // online softmax: row max / rescale / exp / row sum
    float newm[8], alpha[8];
#pragma unroll
    for (int r = 0; r < 8; ++r) {
      float mx = sacc[0][r];
#pragma unroll
      for (int nt = 1; nt < 4; ++nt) mx = fmaxf(mx, sacc[nt][r]);
      for (int o = 8; o >= 1; o >>= 1) mx = fmaxf(mx, __shfl_xor(mx, o, 32));
      newm[r]  = fmaxf(mrow[r], mx);
      alpha[r] = __expf(mrow[r] - newm[r]);
      mrow[r]  = newm[r];
    }
#pragma unroll
    for (int r = 0; r < 8; ++r) {
      float rs = 0.0f;
#pragma unroll
      for (int nt = 0; nt < 4; ++nt) {
        float p = __expf(sacc[nt][r] - newm[r]);
        Ps[w * 16 + half * 8 + r][nt * 16 + l15] = p;   // C-layout store
        rs += p;
      }
      for (int o = 8; o >= 1; o >>= 1) rs += __shfl_xor(rs, o, 32);
      lrow[r] = lrow[r] * alpha[r] + rs;
#pragma unroll
      for (int nt = 0; nt < 4; ++nt) oacc[nt][r] *= alpha[r];
    }
    __syncthreads();

    // O += P @ V  (contraction over s): reload P in A-layout
#pragma unroll
    for (int ks = 0; ks < 16; ++ks) {
      v2f a;
      a[0] = Ps[w * 16 + l15][ks * 4 + half * 2 + 0];
      a[1] = Ps[w * 16 + l15][ks * 4 + half * 2 + 1];
#pragma unroll
      for (int nt = 0; nt < 4; ++nt) {
        v2f bf;
        bf[0] = Vc[ks * 4 + half * 2 + 0][nt * 16 + l15];  // B: K = s, N = d
        bf[1] = Vc[ks * 4 + half * 2 + 1][nt * 16 + l15];
        oacc[nt] = __builtin_amdgcn_wmma_f32_16x16x4_f32(false, a, false, bf,
                                                         (short)0, oacc[nt], false, false);
      }
    }
  }

  // normalize and store (undo head transpose)
#pragma unroll
  for (int nt = 0; nt < 4; ++nt)
#pragma unroll
    for (int r = 0; r < 8; ++r) {
      int row = b * T_C + w * 16 + half * 8 + r;
      int col = h * HD_C + nt * 16 + l15;
      O[(size_t)row * DM_C + col] = oacc[nt][r] / lrow[r];
    }
}

// ---------------------------------------------------------------------------
// y = LayerNorm(a + x) over last dim D=256; one block (256 thr) per row
// ---------------------------------------------------------------------------
__global__ __launch_bounds__(256) void add_ln_kernel(
    const float* __restrict__ A, const float* __restrict__ X,
    const float* __restrict__ g, const float* __restrict__ bta,
    float* __restrict__ Y)
{
  __shared__ float red[8];
  const int row = blockIdx.x;
  const int t = threadIdx.x;
  float v = A[(size_t)row * P_C + t] + X[(size_t)row * P_C + t];

  float s = v;
  for (int o = 16; o >= 1; o >>= 1) s += __shfl_xor(s, o, 32);
  if ((t & 31) == 0) red[t >> 5] = s;
  __syncthreads();
  float tot = 0.0f;
#pragma unroll
  for (int i = 0; i < 8; ++i) tot += red[i];
  float mean = tot * (1.0f / P_C);
  __syncthreads();

  float d = v - mean;
  float s2 = d * d;
  for (int o = 16; o >= 1; o >>= 1) s2 += __shfl_xor(s2, o, 32);
  if ((t & 31) == 0) red[t >> 5] = s2;
  __syncthreads();
  float var = 0.0f;
#pragma unroll
  for (int i = 0; i < 8; ++i) var += red[i];
  var *= (1.0f / P_C);

  Y[(size_t)row * P_C + t] = d * rsqrtf(var + 1e-5f) * g[t] + bta[t];
}

// ---------------------------------------------------------------------------
extern "C" void kernel_launch(void* const* d_in, const int* in_sizes, int n_in,
                              void* d_out, int out_size, void* d_ws, size_t ws_size,
                              hipStream_t stream) {
  (void)in_sizes; (void)n_in; (void)out_size; (void)ws_size;
  const float* src  = (const float*)d_in[0];
  const float* tgt  = (const float*)d_in[1];
  const float* Wq   = (const float*)d_in[2];
  const float* bq   = (const float*)d_in[3];
  const float* Wk   = (const float*)d_in[4];
  const float* bk   = (const float*)d_in[5];
  const float* Wv   = (const float*)d_in[6];
  const float* bv   = (const float*)d_in[7];
  const float* Wo   = (const float*)d_in[8];
  const float* bo   = (const float*)d_in[9];
  const float* ln1g = (const float*)d_in[10];
  const float* ln1b = (const float*)d_in[11];
  const float* W1   = (const float*)d_in[12];
  const float* bf1  = (const float*)d_in[13];
  const float* W2   = (const float*)d_in[14];
  const float* bf2  = (const float*)d_in[15];
  const float* ln3g = (const float*)d_in[16];
  const float* ln3b = (const float*)d_in[17];

  char* ws = (char*)d_ws;
  size_t off = 0;
  auto alloc = [&](size_t bytes) -> void* {
    void* p = ws + off;
    off += (bytes + 255) & ~(size_t)255;
    return p;
  };

  const size_t BT = (size_t)B_C * T_C;  // 1024
  const size_t BS = (size_t)B_C * S_C;  // 32768

  float*  Kp    = (float*) alloc(BS * DM_C * 4);
  float*  Vp    = (float*) alloc(BS * DM_C * 4);
  float*  Qb    = (float*) alloc(BT * DM_C * 4);
  float*  Ab    = (float*) alloc(BT * DM_C * 4);   // attention output
  float*  Aout  = (float*) alloc(BT * P_C * 4);    // out proj
  float*  X1    = (float*) alloc(BT * P_C * 4);
  float*  Hff   = (float*) alloc(BT * DM_C * 4);
  float*  Fout  = (float*) alloc(BT * P_C * 4);
  __bf16* WqTh  = (__bf16*)alloc((size_t)DM_C * P_C * 2);
  __bf16* WqTl  = (__bf16*)alloc((size_t)DM_C * P_C * 2);
  __bf16* WkTh  = (__bf16*)alloc((size_t)DM_C * DM_C * 2);
  __bf16* WkTl  = (__bf16*)alloc((size_t)DM_C * DM_C * 2);
  __bf16* WvTh  = (__bf16*)alloc((size_t)DM_C * DM_C * 2);
  __bf16* WvTl  = (__bf16*)alloc((size_t)DM_C * DM_C * 2);
  __bf16* WoTh  = (__bf16*)alloc((size_t)P_C * DM_C * 2);
  __bf16* WoTl  = (__bf16*)alloc((size_t)P_C * DM_C * 2);
  __bf16* W1Th  = (__bf16*)alloc((size_t)DM_C * P_C * 2);
  __bf16* W1Tl  = (__bf16*)alloc((size_t)DM_C * P_C * 2);
  __bf16* W2Th  = (__bf16*)alloc((size_t)P_C * DM_C * 2);
  __bf16* W2Tl  = (__bf16*)alloc((size_t)P_C * DM_C * 2);

  // 1. repack weights (hi/lo bf16, transposed)
  repack_weight<<<(P_C * DM_C + 255) / 256, 256, 0, stream>>>(Wq, WqTh, WqTl, P_C, DM_C);
  repack_weight<<<(DM_C * DM_C + 255) / 256, 256, 0, stream>>>(Wk, WkTh, WkTl, DM_C, DM_C);
  repack_weight<<<(DM_C * DM_C + 255) / 256, 256, 0, stream>>>(Wv, WvTh, WvTl, DM_C, DM_C);
  repack_weight<<<(DM_C * P_C + 255) / 256, 256, 0, stream>>>(Wo, WoTh, WoTl, DM_C, P_C);
  repack_weight<<<(P_C * DM_C + 255) / 256, 256, 0, stream>>>(W1, W1Th, W1Tl, P_C, DM_C);
  repack_weight<<<(DM_C * P_C + 255) / 256, 256, 0, stream>>>(W2, W2Th, W2Tl, DM_C, P_C);

  // 2. projections
  gemm_bf16x3<false><<<dim3(BT / 64, DM_C / 64), 128, 0, stream>>>(
      tgt, WqTh, WqTl, bq, Qb, (int)BT, DM_C, P_C);
  gemm_bf16x3<false><<<dim3(BS / 64, DM_C / 64), 128, 0, stream>>>(
      src, WkTh, WkTl, bk, Kp, (int)BS, DM_C, DM_C);
  gemm_bf16x3<false><<<dim3(BS / 64, DM_C / 64), 128, 0, stream>>>(
      src, WvTh, WvTl, bv, Vp, (int)BS, DM_C, DM_C);

  // 3. attention (one block per (b,h))
  attn_kernel<<<B_C * H_C, 128, 0, stream>>>(Qb, Kp, Vp, Ab);

  // 4. output projection + residual LN
  gemm_bf16x3<false><<<dim3(BT / 64, P_C / 64), 128, 0, stream>>>(
      Ab, WoTh, WoTl, bo, Aout, (int)BT, P_C, DM_C);
  add_ln_kernel<<<BT, 256, 0, stream>>>(Aout, tgt, ln1g, ln1b, X1);

  // 5. FFN + residual LN
  gemm_bf16x3<true><<<dim3(BT / 64, DM_C / 64), 128, 0, stream>>>(
      X1, W1Th, W1Tl, bf1, Hff, (int)BT, DM_C, P_C);
  gemm_bf16x3<false><<<dim3(BT / 64, P_C / 64), 128, 0, stream>>>(
      Hff, W2Th, W2Tl, bf2, Fout, (int)BT, P_C, DM_C);
  add_ln_kernel<<<BT, 256, 0, stream>>>(Fout, X1, ln3g, ln3b, (float*)d_out);
}
